// PDN_dense_add_skip_3bro_pre_batch_act_max_test_38010460569943
// MI455X (gfx1250) — matrix operand order repaired and checked by reading, not compiled, not run on hardware
//
#include <hip/hip_runtime.h>
#include <hip/hip_bf16.h>

typedef __attribute__((ext_vector_type(16))) _Float16 v16h;
typedef __attribute__((ext_vector_type(8)))  float    v8f;

#define CDIM 128
#define GDIM 2048
#define NCDIM 10
#define EPSV 1e-5f

// ---------------------------------------------------------------------------
// Edge MLP: w[e] = sigmoid( relu(ea @ W1 + b1) @ W2 + b2 )
// ---------------------------------------------------------------------------
__global__ __launch_bounds__(256)
void edge_mlp_kernel(const float* __restrict__ EA, const float* __restrict__ W1,
                     const float* __restrict__ b1, const float* __restrict__ W2,
                     const float* __restrict__ b2, float* __restrict__ w, int E) {
  __shared__ float sW1[256];
  __shared__ float sb1[16];
  __shared__ float sW2[16];
  __shared__ float sb2;
  int t = threadIdx.x;
  sW1[t] = W1[t];
  if (t < 16) { sb1[t] = b1[t]; sW2[t] = W2[t]; }
  if (t == 0) sb2 = b2[0];
  __syncthreads();
  int e = blockIdx.x * 256 + t;
  if (e >= E) return;
  float a[16];
  const float4* ap = (const float4*)(EA + (size_t)e * 16);
#pragma unroll
  for (int q = 0; q < 4; ++q) {
    float4 v = ap[q];
    a[q*4+0] = v.x; a[q*4+1] = v.y; a[q*4+2] = v.z; a[q*4+3] = v.w;
  }
  float z = sb2;
#pragma unroll
  for (int j = 0; j < 16; ++j) {
    float h = sb1[j];
#pragma unroll
    for (int i = 0; i < 16; ++i) h += a[i] * sW1[i*16 + j];
    h = fmaxf(h, 0.f);
    z += h * sW2[j];
  }
  w[e] = 1.f / (1.f + __expf(-z));
}

// ---------------------------------------------------------------------------
// Small elementwise helpers
// ---------------------------------------------------------------------------
__global__ __launch_bounds__(256)
void fill_kernel(float* __restrict__ p, float v, int n) {
  int i = blockIdx.x * 256 + threadIdx.x;
  if (i < n) p[i] = v;
}

__global__ __launch_bounds__(256)
void deg_scatter_kernel(const int* __restrict__ dst, const float* __restrict__ w,
                        float* __restrict__ deg, int E) {
  int e = blockIdx.x * 256 + threadIdx.x;
  if (e < E) atomicAdd(&deg[dst[e]], w[e]);
}

__global__ __launch_bounds__(256)
void rsqrt_kernel(float* __restrict__ p, int n) {
  int i = blockIdx.x * 256 + threadIdx.x;
  if (i < n) p[i] = rsqrtf(p[i]);   // deg >= 1 always (self loop)
}

__global__ __launch_bounds__(256)
void edge_norm_kernel(const int* __restrict__ src, const int* __restrict__ dst,
                      const float* __restrict__ w, const float* __restrict__ dinv,
                      float* __restrict__ nrm, int E) {
  int e = blockIdx.x * 256 + threadIdx.x;
  if (e < E) nrm[e] = dinv[src[e]] * w[e] * dinv[dst[e]];
}

// ---------------------------------------------------------------------------
// Column stats for BatchNorm (batch statistics, biased var)
// STATS layout: [0:128)=sum [128:256)=sumsq [256:384)=mean [384:512)=rstd
// ---------------------------------------------------------------------------
__global__ __launch_bounds__(256)
void col_stats_kernel(const float* __restrict__ X, float* __restrict__ stats, int N) {
  int t = threadIdx.x;
  int c = t & 127;
  int ro = t >> 7;                 // 0 or 1
  float s = 0.f, s2 = 0.f;
  for (int r = blockIdx.x * 2 + ro; r < N; r += gridDim.x * 2) {
    float v = X[(size_t)r * CDIM + c];
    s += v; s2 += v * v;
  }
  __shared__ float ls[128], ls2[128];
  if (ro == 0) { ls[c] = s; ls2[c] = s2; }
  __syncthreads();
  if (ro == 1) { ls[c] += s; ls2[c] += s2; }
  __syncthreads();
  if (ro == 0) {
    atomicAdd(&stats[c],       ls[c]);
    atomicAdd(&stats[128 + c], ls2[c]);
  }
}

__global__ __launch_bounds__(128)
void stats_finalize_kernel(float* __restrict__ stats, int N) {
  int c = threadIdx.x;
  float inv = 1.f / (float)N;
  float mu = stats[c] * inv;
  float var = stats[128 + c] * inv - mu * mu;
  stats[256 + c] = mu;
  stats[384 + c] = rsqrtf(var + EPSV);
}

// ---------------------------------------------------------------------------
// WMMA GEMM: H = act(X) @ Wn   ([N,128] x [128,128])
// USEBN compile-time: act(v) = max(0, v*scale[k] + shift[k]) with scale/shift
// staged in LDS (branch-free inner loop). A loaded as float4 runs, B as two 16B
// LDS vectors matching the ISA f16 fragment layouts. 4 waves/block, each wave
// a 16x128 strip: 8 accum tiles, K=128 in 4 steps of v_wmma_f32_16x16x32_f16.
// ---------------------------------------------------------------------------
template<bool USEBN>
__global__ __launch_bounds__(128)
void gemm_wmma_kernel(const float* __restrict__ X, const float* __restrict__ Wn,
                      const float* __restrict__ stats, float* __restrict__ H, int N) {
  __shared__ __align__(32) _Float16 WT[CDIM * CDIM];   // 32 KB, [col][k]
  __shared__ __align__(32) float sSc[CDIM];
  __shared__ __align__(32) float sSh[CDIM];
  int t = threadIdx.x;
  if (USEBN && t < CDIM) {
    float mu = stats[256 + t];
    float rs = stats[384 + t];
    sSc[t] = rs;
    sSh[t] = -mu * rs;
  }
  for (int i = t; i < CDIM * CDIM; i += 128) {
    int k = i >> 7, n = i & 127;
    WT[n * CDIM + k] = (_Float16)Wn[k * CDIM + n];
  }
  __syncthreads();

  int wave = t >> 5;
  int lane = t & 31;
  int m = lane & 15;
  int half = lane >> 4;
  int rowBase = (blockIdx.x * 4 + wave) * 16;
  int row = rowBase + m;
  int rowc = row < N ? row : N - 1;
  const float* xrow = X + (size_t)rowc * CDIM;

  v8f zero = {0.f, 0.f, 0.f, 0.f, 0.f, 0.f, 0.f, 0.f};
  v8f acc[8];
#pragma unroll
  for (int nt = 0; nt < 8; ++nt) acc[nt] = zero;

  int kbA = half * 8;    // A frag: this lane owns K kb..kb+7 and kb+16..kb+23
  int kbB = half * 8;    // B frag: same as A, K kb..kb+7 and kb+16..kb+23

#pragma unroll
  for (int ks = 0; ks < 4; ++ks) {
    int kA = ks * 32 + kbA;
    float4 xa0 = *(const float4*)(xrow + kA);
    float4 xa1 = *(const float4*)(xrow + kA + 4);
    float4 xb0 = *(const float4*)(xrow + kA + 16);
    float4 xb1 = *(const float4*)(xrow + kA + 20);
    float va[8] = {xa0.x, xa0.y, xa0.z, xa0.w, xa1.x, xa1.y, xa1.z, xa1.w};
    float vb[8] = {xb0.x, xb0.y, xb0.z, xb0.w, xb1.x, xb1.y, xb1.z, xb1.w};
    if (USEBN) {
      float4 sa0 = *(const float4*)(sSc + kA);
      float4 sa1 = *(const float4*)(sSc + kA + 4);
      float4 sb0 = *(const float4*)(sSc + kA + 16);
      float4 sb1 = *(const float4*)(sSc + kA + 20);
      float4 ha0 = *(const float4*)(sSh + kA);
      float4 ha1 = *(const float4*)(sSh + kA + 4);
      float4 hb0 = *(const float4*)(sSh + kA + 16);
      float4 hb1 = *(const float4*)(sSh + kA + 20);
      float sca[8] = {sa0.x, sa0.y, sa0.z, sa0.w, sa1.x, sa1.y, sa1.z, sa1.w};
      float scb[8] = {sb0.x, sb0.y, sb0.z, sb0.w, sb1.x, sb1.y, sb1.z, sb1.w};
      float sha[8] = {ha0.x, ha0.y, ha0.z, ha0.w, ha1.x, ha1.y, ha1.z, ha1.w};
      float shb[8] = {hb0.x, hb0.y, hb0.z, hb0.w, hb1.x, hb1.y, hb1.z, hb1.w};
#pragma unroll
      for (int j = 0; j < 8; ++j) {
        va[j] = fmaxf(fmaf(va[j], sca[j], sha[j]), 0.f);
        vb[j] = fmaxf(fmaf(vb[j], scb[j], shb[j]), 0.f);
      }
    }
    v16h a;
#pragma unroll
    for (int j = 0; j < 8; ++j) {
      a[j]     = (_Float16)va[j];
      a[8 + j] = (_Float16)vb[j];
    }
#pragma unroll
    for (int nt = 0; nt < 8; ++nt) {
      int col = nt * 16 + m;
      typedef __attribute__((ext_vector_type(8))) _Float16 v8h;
      v8h b0 = *(const v8h*)&WT[col * CDIM + ks * 32 + kbB];
      v8h b1 = *(const v8h*)&WT[col * CDIM + ks * 32 + kbB + 16];
      v16h b = __builtin_shufflevector(b0, b1, 0, 1, 2, 3, 4, 5, 6, 7, 8, 9, 10, 11, 12, 13, 14, 15);
      acc[nt] = __builtin_amdgcn_wmma_f32_16x16x32_f16(
          false, a, false, b, (short)0, acc[nt], false, false);
    }
  }
  // C/D layout: VGPR j -> lanes 0-15: M=j, N=lane; lanes 16-31: M=j+8, N=lane-16
#pragma unroll
  for (int nt = 0; nt < 8; ++nt) {
    int col = nt * 16 + m;
#pragma unroll
    for (int j = 0; j < 8; ++j) {
      int r = rowBase + j + half * 8;
      if (r < N) H[(size_t)r * CDIM + col] = acc[nt][j];
    }
  }
}

// ---------------------------------------------------------------------------
// out = bias + dinv^2 * h   (self-loop term fused with output init)
// ---------------------------------------------------------------------------
__global__ __launch_bounds__(256)
void out_init_kernel(const float* __restrict__ H, const float* __restrict__ dinv,
                     const float* __restrict__ bias, float* __restrict__ O, int N) {
  unsigned tid = blockIdx.x * 256 + threadIdx.x;
  if (tid >= (unsigned)N * 32u) return;
  int i = tid >> 5;
  int c4 = (tid & 31) * 4;
  float dn = dinv[i];
  float s = dn * dn;
  float4 h = *(const float4*)(H + (size_t)i * CDIM + c4);
  float4 b = *(const float4*)(bias + c4);
  float4 o;
  o.x = b.x + s * h.x; o.y = b.y + s * h.y;
  o.z = b.z + s * h.z; o.w = b.w + s * h.w;
  *(float4*)(O + (size_t)i * CDIM + c4) = o;
}

// ---------------------------------------------------------------------------
// Edge scatter: out[dst] += norm[e] * h[src]; one wave per edge, float4/lane.
// Gathers & atomics resolve in the 192MB L2 (h + out both L2-resident).
// ---------------------------------------------------------------------------
__global__ __launch_bounds__(256)
void scatter_kernel(const int* __restrict__ src, const int* __restrict__ dst,
                    const float* __restrict__ nrm, const float* __restrict__ H,
                    float* __restrict__ O, int E) {
  unsigned tid = blockIdx.x * 256 + threadIdx.x;
  if (tid >= (unsigned)E * 32u) return;
  int e = tid >> 5;
  int c4 = (tid & 31) * 4;
  float nv = nrm[e];
  const float4 h = *(const float4*)(H + (size_t)src[e] * CDIM + c4);
  float* op = O + (size_t)dst[e] * CDIM + c4;
  atomicAdd(op + 0, nv * h.x);
  atomicAdd(op + 1, nv * h.y);
  atomicAdd(op + 2, nv * h.z);
  atomicAdd(op + 3, nv * h.w);
}

// ---------------------------------------------------------------------------
// Skip combinations
// ---------------------------------------------------------------------------
__global__ __launch_bounds__(256)
void add2_kernel(float* __restrict__ d, const float* __restrict__ a,
                 const float* __restrict__ b, int n) {
  int i = blockIdx.x * 256 + threadIdx.x;
  if (i < n) d[i] = a[i] + b[i];
}

__global__ __launch_bounds__(256)
void add3_kernel(float* __restrict__ d, const float* __restrict__ a,
                 const float* __restrict__ b, const float* __restrict__ c, int n) {
  int i = blockIdx.x * 256 + threadIdx.x;
  if (i < n) d[i] = a[i] + b[i] + c[i];
}

__global__ __launch_bounds__(256)
void relu_add4_kernel(float* __restrict__ d, const float* __restrict__ a,
                      const float* __restrict__ b, const float* __restrict__ c,
                      const float* __restrict__ e, int n) {
  int i = blockIdx.x * 256 + threadIdx.x;
  if (i < n) d[i] = fmaxf(a[i] + b[i] + c[i] + e[i], 0.f);
}

// ---------------------------------------------------------------------------
// Global max pool per graph. x3 >= 0 post-ReLU, so uint bit pattern is
// order-preserving -> atomicMax on uint, pool pre-zeroed.
// ---------------------------------------------------------------------------
__global__ __launch_bounds__(256)
void pool_max_kernel(const float* __restrict__ X3, const int* __restrict__ batch,
                     unsigned* __restrict__ P, int N) {
  unsigned tid = blockIdx.x * 256 + threadIdx.x;
  if (tid >= (unsigned)N * 32u) return;
  int i = tid >> 5;
  int c4 = (tid & 31) * 4;
  int g = batch[i];
  float4 v = *(const float4*)(X3 + (size_t)i * CDIM + c4);
  unsigned* pp = P + (size_t)g * CDIM + c4;
  atomicMax(pp + 0, __float_as_uint(v.x));
  atomicMax(pp + 1, __float_as_uint(v.y));
  atomicMax(pp + 2, __float_as_uint(v.z));
  atomicMax(pp + 3, __float_as_uint(v.w));
}

__global__ __launch_bounds__(256)
void final_linear_kernel(const float* __restrict__ P, const float* __restrict__ W,
                         const float* __restrict__ b, float* __restrict__ out) {
  int tid = blockIdx.x * 256 + threadIdx.x;
  if (tid >= GDIM * NCDIM) return;
  int g = tid / NCDIM, nc = tid % NCDIM;
  const float* pr = P + (size_t)g * CDIM;
  float s = b[nc];
#pragma unroll 8
  for (int c = 0; c < CDIM; ++c) s += pr[c] * W[c * NCDIM + nc];
  out[tid] = s;
}

// ---------------------------------------------------------------------------
// Host orchestration
// ---------------------------------------------------------------------------
extern "C" void kernel_launch(void* const* d_in, const int* in_sizes, int n_in,
                              void* d_out, int out_size, void* d_ws, size_t ws_size,
                              hipStream_t stream) {
  const float* x_in      = (const float*)d_in[0];
  const int*   ei        = (const int*)  d_in[1];
  const int*   batch     = (const int*)  d_in[2];
  const float* edge_attr = (const float*)d_in[4];
  const int N = in_sizes[0] / CDIM;
  const int E = in_sizes[1] / 2;
  const int* srcI = ei;
  const int* dstI = ei + E;

  // conv params: 5..10 = c1, 11..16 = h1[3], 17..22 = h2[3], 23..28 = h3[3]
  const float* c1p[6]; for (int i = 0; i < 6; ++i) c1p[i] = (const float*)d_in[5 + i];
  const float* h1p[6]; for (int i = 0; i < 6; ++i) h1p[i] = (const float*)d_in[11 + i];
  const float* h2p[6]; for (int i = 0; i < 6; ++i) h2p[i] = (const float*)d_in[17 + i];
  const float* h3p[6]; for (int i = 0; i < 6; ++i) h3p[i] = (const float*)d_in[23 + i];
  const float* lin_W = (const float*)d_in[29];
  const float* lin_b = (const float*)d_in[30];
  // per-layer element strides for stacked params: Wn,eW1,eb1,eW2,eb2,bias
  const size_t pstride[6] = {CDIM * CDIM, 16 * 16, 16, 16, 1, CDIM};

  char* ws = (char*)d_ws;
  size_t off = 0;
  auto alloc = [&](size_t bytes) -> void* {
    void* p = ws + off;
    off += (bytes + 255) & ~(size_t)255;
    return p;
  };
  size_t szNC = (size_t)N * CDIM * sizeof(float);
  float* P0   = (float*)alloc(szNC);
  float* P1   = (float*)alloc(szNC);
  float* P2   = (float*)alloc(szNC);
  float* P3   = (float*)alloc(szNC);
  float* P4   = (float*)alloc(szNC);
  float* Hbuf = (float*)alloc(szNC);
  float* Wbuf = (float*)alloc((size_t)E * 4);
  float* Nrm  = (float*)alloc((size_t)E * 4);
  float* Dinv = (float*)alloc((size_t)N * 4);
  float* Stat = (float*)alloc(512 * 4);
  unsigned* Pool = (unsigned*)alloc((size_t)GDIM * CDIM * 4);

  const unsigned BLK = 256;
  unsigned gE   = (E + BLK - 1) / BLK;
  unsigned gN   = (N + BLK - 1) / BLK;
  unsigned gN32 = ((unsigned)N * 32u + BLK - 1) / BLK;
  unsigned gE32 = ((unsigned)E * 32u + BLK - 1) / BLK;
  unsigned gNC  = ((unsigned)N * CDIM + BLK - 1) / BLK;

  auto conv = [&](const float* xin, float* out, const float* p[6], int layer, int useBN) {
    const float* Wn  = p[0] + (size_t)layer * pstride[0];
    const float* eW1 = p[1] + (size_t)layer * pstride[1];
    const float* eb1 = p[2] + (size_t)layer * pstride[2];
    const float* eW2 = p[3] + (size_t)layer * pstride[3];
    const float* eb2 = p[4] + (size_t)layer * pstride[4];
    const float* bia = p[5] + (size_t)layer * pstride[5];
    edge_mlp_kernel<<<gE, BLK, 0, stream>>>(edge_attr, eW1, eb1, eW2, eb2, Wbuf, E);
    fill_kernel<<<gN, BLK, 0, stream>>>(Dinv, 1.0f, N);                 // self-loop weight
    deg_scatter_kernel<<<gE, BLK, 0, stream>>>(dstI, Wbuf, Dinv, E);
    rsqrt_kernel<<<gN, BLK, 0, stream>>>(Dinv, N);
    edge_norm_kernel<<<gE, BLK, 0, stream>>>(srcI, dstI, Wbuf, Dinv, Nrm, E);
    if (useBN) {
      fill_kernel<<<1, BLK, 0, stream>>>(Stat, 0.f, 256);
      col_stats_kernel<<<512, BLK, 0, stream>>>(xin, Stat, N);
      stats_finalize_kernel<<<1, 128, 0, stream>>>(Stat, N);
      gemm_wmma_kernel<true><<<(N + 63) / 64, 128, 0, stream>>>(xin, Wn, Stat, Hbuf, N);
    } else {
      gemm_wmma_kernel<false><<<(N + 63) / 64, 128, 0, stream>>>(xin, Wn, Stat, Hbuf, N);
    }
    out_init_kernel<<<gN32, BLK, 0, stream>>>(Hbuf, Dinv, bia, out, N);
    scatter_kernel<<<gE32, BLK, 0, stream>>>(srcI, dstI, Nrm, Hbuf, out, E);
  };

  int nNC = N * CDIM;
  // c1 (no BN): x0 = x = P0
  conv(x_in, P0, c1p, 0, 0);
  // block 1
  conv(P0, P1, h1p, 0, 1);
  conv(P1, P2, h1p, 1, 1);
  conv(P2, P1, h1p, 2, 1);
  add2_kernel<<<gNC, BLK, 0, stream>>>(P3, P1, P0, nNC);        // x1 = P3
  // block 2
  conv(P3, P1, h2p, 0, 1);
  conv(P1, P2, h2p, 1, 1);
  conv(P2, P1, h2p, 2, 1);
  add3_kernel<<<gNC, BLK, 0, stream>>>(P4, P1, P0, P3, nNC);    // x2 = P4
  // block 3
  conv(P4, P1, h3p, 0, 1);
  conv(P1, P2, h3p, 1, 1);
  conv(P2, P1, h3p, 2, 1);
  relu_add4_kernel<<<gNC, BLK, 0, stream>>>(P2, P1, P0, P3, P4, nNC);   // x3 = P2
  // pool + classifier
  fill_kernel<<<(GDIM * CDIM + BLK - 1) / BLK, BLK, 0, stream>>>((float*)Pool, 0.f, GDIM * CDIM);
  pool_max_kernel<<<gN32, BLK, 0, stream>>>(P2, batch, Pool, N);
  final_linear_kernel<<<(GDIM * NCDIM + BLK - 1) / BLK, BLK, 0, stream>>>(
      (const float*)Pool, lin_W, lin_b, (float*)d_out);
}
